// PrEL2P_51505247814360
// MI455X (gfx1250) — compile-verified, matrix operand order checked by reading, not run on hardware
//
#include <hip/hip_runtime.h>
#include <hip/hip_bf16.h>
#include <math.h>

#define Lnum 12
#define Hn 12
#define Cdim 768
#define Ntok 197
#define Bsz 32
#define NSTALE 56
#define CLS 100
#define NKEYPAD 224

// LDS tile geometry
#define ASTRIDE 36                 // A tile: 32 f32 cols + pad
#define ATILE (32 * ASTRIDE)       // f32 elements
#define BSTRIDEH 40                // B^T tile: 32 bf16 cols + pad (80B, 16B-aligned)
#define BTILEH (256 * BSTRIDEH)    // bf16 elements

typedef __attribute__((ext_vector_type(16))) __bf16 v16bf;
typedef __attribute__((ext_vector_type(8)))  __bf16 v8bf;
typedef __attribute__((ext_vector_type(8)))  float  v8f;

#if defined(__has_builtin)
# if __has_builtin(__builtin_amdgcn_global_load_async_to_lds_b128)
#  define HAVE_ASYNC 1
# endif
# if __has_builtin(__builtin_amdgcn_s_wait_asynccnt)
#  define HAVE_WAITASYNC 1
# endif
#endif

// param 1: int4 addrspace(1)* (global src); param 2: LDS dst (per hipcc diag)
typedef int v4i_ __attribute__((vector_size(16)));
typedef __attribute__((address_space(1))) v4i_ GVec;
typedef __attribute__((address_space(3))) v4i_ LVec;

static __device__ __forceinline__ __bf16 f2bf(float f) { return (__bf16)f; }

// 16-byte global->LDS copy: async on CDNA5 (ASYNCcnt), sync fallback otherwise.
static __device__ __forceinline__ void cp16_g2l(const void* g, void* l) {
#if HAVE_ASYNC
  __builtin_amdgcn_global_load_async_to_lds_b128((GVec*)g, (LVec*)l, 0, 0);
#else
  *(v4i_*)l = *(const v4i_*)g;
#endif
}

template <int N>
static __device__ __forceinline__ void wait_async() {
#if HAVE_ASYNC
# if HAVE_WAITASYNC
  __builtin_amdgcn_s_wait_asynccnt(N);
# else
  asm volatile("s_wait_asynccnt %0" ::"n"(N) : "memory");
# endif
#endif
}

// ---------------------------------------------------------------------------
// One-time weight transpose+convert: Wt[n][k] = bf16(W[k][n]).
// blockIdx = (N/32, K/32, layer); classic 32x33 LDS tile, both sides coalesced.
// ---------------------------------------------------------------------------
__global__ __launch_bounds__(256) void wt_bf16_kernel(
    const float* __restrict__ W, __bf16* __restrict__ Wt, int K, int N)
{
  __shared__ float tile[32][33];
  const long moff = (long)blockIdx.z * K * N;
  const int tx = threadIdx.x & 31, ty = threadIdx.x >> 5;   // 32 x 8
  const int n0 = blockIdx.x * 32, k0 = blockIdx.y * 32;
  #pragma unroll
  for (int i = 0; i < 32; i += 8)
    tile[ty + i][tx] = W[moff + (long)(k0 + ty + i) * N + n0 + tx];
  __syncthreads();
  #pragma unroll
  for (int i = 0; i < 32; i += 8)
    Wt[moff + (long)(n0 + ty + i) * K + k0 + tx] = f2bf(tile[tx][ty + i]);
}

// ---------------------------------------------------------------------------
// WMMA GEMM: C[M,N] = act(A[M,K] @ W[K,N] + bias) (+ residual)
// A: f32 [M,K]. Wt: bf16 [N,K] (pre-transposed). Block 256 thr = 8 waves,
// 2(M) x 4(N) wave tiles of 16Mx64N -> block tile 32M x 256N.
// K in 32-slabs, double-buffered async->LDS: per thread exactly 5 b128
// (1 A f32 chunk + 4 B^T bf16 chunks) => s_wait_asynccnt 5 is exact.
// B fragments: contiguous 16B ds_load_b128, zero cvt in inner loop.
// ---------------------------------------------------------------------------
__global__ __launch_bounds__(256) void gemm_wmma_kernel(
    const float* __restrict__ A, const __bf16* __restrict__ Wt,
    const float* __restrict__ bias, const float* __restrict__ Res,
    float* __restrict__ C, int M, int N, int K, int act)
{
  extern __shared__ char smemraw[];
  float*  As = (float*)smemraw;                            // 2 x ATILE f32
  __bf16* Bs = (__bf16*)(smemraw + 2 * ATILE * sizeof(float)); // 2 x BTILEH bf16

  const int tid = threadIdx.x;
  const int lane = tid & 31;
  const int wave = tid >> 5;
  const int wm = wave >> 2;        // 0..1
  const int wn = wave & 3;         // 0..3
  const int half = lane >> 4;
  const int l15 = lane & 15;
  const int kb = half * 8;         // K sub-offset per ISA 16-bit A/B layout
  const int m0base = blockIdx.y * 32;
  const int n0base = blockIdx.x * 256;

  const int arow = tid >> 3, acol = (tid & 7) * 4;         // A: 1 chunk/thread
  const long abase = (long)(m0base + arow) * K + acol;

  auto stage = [&](int buf, int kk) {
    cp16_g2l(A + abase + kk, As + buf * ATILE + arow * ASTRIDE + acol);
    #pragma unroll
    for (int i = 0; i < 4; ++i) {                          // B^T: 4 chunks/thread
      const int c = tid + i * 256;                         // 0..1023
      const int r = c >> 2, coff = (c & 3) * 8;            // row n, bf16 col offset
      cp16_g2l(Wt + (long)(n0base + r) * K + kk + coff,
               Bs + buf * BTILEH + r * BSTRIDEH + coff);
    }
  };

  v8f acc0 = {}, acc1 = {}, acc2 = {}, acc3 = {};
  const int nsteps = K >> 5;

  stage(0, 0);
  for (int s = 0; s < nsteps; ++s) {
    const int buf = s & 1;
    if (s + 1 < nsteps) { stage(buf ^ 1, (s + 1) << 5); wait_async<5>(); }
    else                { wait_async<0>(); }
    __syncthreads();

    const float*  Ac = As + buf * ATILE;
    const __bf16* Bc = Bs + buf * BTILEH;

    // A fragment: row (wm*16+l15), K = kb+[0..7] and kb+16+[0..7] (8 cvts)
    v16bf af;
    const float* ar = Ac + (wm * 16 + l15) * ASTRIDE + kb;
    #pragma unroll
    for (int e = 0; e < 8; ++e) { af[e] = f2bf(ar[e]); af[8 + e] = f2bf(ar[16 + e]); }

    // 4 B fragments: two contiguous 16B LDS vector loads each, no cvt
    #pragma unroll
    for (int t = 0; t < 4; ++t) {
      const __bf16* br = Bc + (wn * 64 + t * 16 + l15) * BSTRIDEH + kb;
      const v8bf lo = *(const v8bf*)br;
      const v8bf hi = *(const v8bf*)(br + 16);
      v16bf bf;
      #pragma unroll
      for (int e = 0; e < 8; ++e) { bf[e] = lo[e]; bf[8 + e] = hi[e]; }
      v8f* accp = (t == 0) ? &acc0 : (t == 1) ? &acc1 : (t == 2) ? &acc2 : &acc3;
      *accp = __builtin_amdgcn_wmma_f32_16x16x32_bf16(false, af, false, bf,
                                                      (short)0, *accp, false, false);
    }
    __syncthreads();
  }

  #pragma unroll
  for (int t = 0; t < 4; ++t) {
    const v8f acc = (t == 0) ? acc0 : (t == 1) ? acc1 : (t == 2) ? acc2 : acc3;
    const int col = n0base + wn * 64 + t * 16 + l15;
    const float bb = bias[col];
    #pragma unroll
    for (int v = 0; v < 8; ++v) {
      const int r = m0base + wm * 16 + v + half * 8;
      float val = acc[v] + bb;
      if (act == 1) val = 0.5f * val * (1.0f + erff(val * 0.70710678118654752f));
      if (Res) val += Res[(long)r * N + col];
      C[(long)r * N + col] = val;
    }
  }
}

// ---------------------------------------------------------------------------
// Attention: one block per (query-tile, head, batch). 128 thr = 4 waves.
// ---------------------------------------------------------------------------
__global__ __launch_bounds__(128) void attn_wmma_kernel(
    const float* __restrict__ qkv, float* __restrict__ o)
{
  const int qt  = blockIdx.x;
  const int hh  = blockIdx.y;
  const int b   = blockIdx.z;
  const int tid = threadIdx.x;
  const int lane = tid & 31;
  const int wave = tid >> 5;
  const int l15 = lane & 15;
  const int half = lane >> 4;
  const int kb = half * 8;

  __shared__ __bf16 Vt[64 * NKEYPAD];
  __shared__ float  Sb[16 * NKEYPAD];
  __shared__ __bf16 Pa[16 * NKEYPAD];

  const long qb = (long)b * Ntok * 2304;

  for (int idx = tid; idx < Ntok * 64; idx += 128) {
    const int n = idx >> 6, d = idx & 63;
    Vt[d * NKEYPAD + n] = f2bf(qkv[qb + (long)n * 2304 + 1536 + hh * 64 + d]);
  }
  for (int idx = tid; idx < 64 * (NKEYPAD - Ntok); idx += 128) {
    const int d = idx / (NKEYPAD - Ntok);
    const int n = Ntok + idx % (NKEYPAD - Ntok);
    Vt[d * NKEYPAD + n] = f2bf(0.f);
  }

  const int qrow = qt * 16 + l15;
  const bool qv = qrow < Ntok;
  v16bf aq[2];
  #pragma unroll
  for (int s = 0; s < 2; ++s) {
    const int kk = s * 32 + kb;
    #pragma unroll
    for (int e = 0; e < 8; ++e) {
      const float f0 = qv ? qkv[qb + (long)qrow * 2304 + hh * 64 + kk + e] : 0.f;
      const float f1 = qv ? qkv[qb + (long)qrow * 2304 + hh * 64 + kk + 16 + e] : 0.f;
      aq[s][e] = f2bf(f0); aq[s][8 + e] = f2bf(f1);
    }
  }
  __syncthreads();

  for (int j = wave; j < 13; j += 4) {
    const int key = j * 16 + l15;
    const bool kv = key < Ntok;
    v8f c = {};
    #pragma unroll
    for (int s = 0; s < 2; ++s) {
      const int kk = s * 32 + kb;
      v16bf bk;
      #pragma unroll
      for (int e = 0; e < 8; ++e) {
        const float f0 = kv ? qkv[qb + (long)key * 2304 + 768 + hh * 64 + kk + e] : 0.f;
        const float f1 = kv ? qkv[qb + (long)key * 2304 + 768 + hh * 64 + kk + 16 + e] : 0.f;
        bk[e] = f2bf(f0); bk[8 + e] = f2bf(f1);
      }
      c = __builtin_amdgcn_wmma_f32_16x16x32_bf16(false, aq[s], false, bk,
                                                  (short)0, c, false, false);
    }
    #pragma unroll
    for (int v = 0; v < 8; ++v)
      Sb[(v + half * 8) * NKEYPAD + j * 16 + l15] = c[v] * 0.125f;
  }
  __syncthreads();

  if (tid < 16) {
    const int token = qt * 16 + tid;
    if (token < Ntok) {
      float m = -3.0e38f;
      for (int cc = 0; cc < Ntok; ++cc) m = fmaxf(m, Sb[tid * NKEYPAD + cc]);
      float sum = 0.f;
      for (int cc = 0; cc < Ntok; ++cc) {
        const float e = __expf(Sb[tid * NKEYPAD + cc] - m);
        sum += e; Sb[tid * NKEYPAD + cc] = e;
      }
      const float inv = 1.0f / sum;
      for (int cc = 0; cc < Ntok; ++cc)
        Pa[tid * NKEYPAD + cc] = f2bf(Sb[tid * NKEYPAD + cc] * inv);
      for (int cc = Ntok; cc < NKEYPAD; ++cc) Pa[tid * NKEYPAD + cc] = f2bf(0.f);
    } else {
      for (int cc = 0; cc < NKEYPAD; ++cc) Pa[tid * NKEYPAD + cc] = f2bf(0.f);
    }
  }
  __syncthreads();

  {
    const int nt = wave;
    v8f c = {};
    for (int kk = 0; kk < NKEYPAD; kk += 32) {
      v16bf ap, bp;
      const __bf16* par = &Pa[l15 * NKEYPAD + kk + kb];
      #pragma unroll
      for (int e = 0; e < 8; ++e) { ap[e] = par[e]; ap[8 + e] = par[16 + e]; }
      const __bf16* vbr = &Vt[(nt * 16 + l15) * NKEYPAD + kk + kb];
      #pragma unroll
      for (int e = 0; e < 8; ++e) { bp[e] = vbr[e]; bp[8 + e] = vbr[16 + e]; }
      c = __builtin_amdgcn_wmma_f32_16x16x32_bf16(false, ap, false, bp,
                                                  (short)0, c, false, false);
    }
    #pragma unroll
    for (int v = 0; v < 8; ++v) {
      const int token = qt * 16 + v + half * 8;
      if (token < Ntok)
        o[(long)(b * Ntok + token) * Cdim + hh * 64 + nt * 16 + l15] = c[v];
    }
  }
}

// ---------------------------------------------------------------------------
__global__ __launch_bounds__(256) void ln_kernel(
    const float* __restrict__ in, long istride,
    const float* __restrict__ w, const float* __restrict__ bb,
    float* __restrict__ out, long ostride)
{
  const long base = (long)blockIdx.x * istride;
  const int tid = threadIdx.x;
  __shared__ float red[256];
  const float x0 = in[base + tid], x1 = in[base + tid + 256], x2 = in[base + tid + 512];
  red[tid] = x0 + x1 + x2;
  __syncthreads();
  for (int s = 128; s > 0; s >>= 1) { if (tid < s) red[tid] += red[tid + s]; __syncthreads(); }
  const float mean = red[0] * (1.0f / 768.0f);
  __syncthreads();
  const float d0 = x0 - mean, d1 = x1 - mean, d2 = x2 - mean;
  red[tid] = d0 * d0 + d1 * d1 + d2 * d2;
  __syncthreads();
  for (int s = 128; s > 0; s >>= 1) { if (tid < s) red[tid] += red[tid + s]; __syncthreads(); }
  const float rstd = rsqrtf(red[0] * (1.0f / 768.0f) + 1e-6f);
  const long ob = (long)blockIdx.x * ostride;
  out[ob + tid]       = d0 * rstd * w[tid]       + bb[tid];
  out[ob + tid + 256] = d1 * rstd * w[tid + 256] + bb[tid + 256];
  out[ob + tid + 512] = d2 * rstd * w[tid + 512] + bb[tid + 512];
}

// ---------------------------------------------------------------------------
__global__ __launch_bounds__(256) void cls_imp_kernel(
    const float* __restrict__ qkv, float* __restrict__ imp)
{
  const int b = blockIdx.x, tid = threadIdx.x;
  __shared__ float q0s[64];
  __shared__ float red[256];
  const long qb = (long)b * Ntok * 2304;
  float acc = 0.f;
  for (int hh = 0; hh < Hn; ++hh) {
    if (tid < 64) q0s[tid] = qkv[qb + hh * 64 + tid];
    __syncthreads();
    float s = -3.0e38f;
    if (tid < Ntok) {
      float dot = 0.f;
      const float* kp = &qkv[qb + (long)tid * 2304 + 768 + hh * 64];
      #pragma unroll 8
      for (int d = 0; d < 64; ++d) dot += q0s[d] * kp[d];
      s = dot * 0.125f;
    }
    red[tid] = s;
    __syncthreads();
    for (int st = 128; st > 0; st >>= 1) { if (tid < st) red[tid] = fmaxf(red[tid], red[tid + st]); __syncthreads(); }
    const float m = red[0];
    __syncthreads();
    const float e = (tid < Ntok) ? __expf(s - m) : 0.f;
    red[tid] = e;
    __syncthreads();
    for (int st = 128; st > 0; st >>= 1) { if (tid < st) red[tid] += red[tid + st]; __syncthreads(); }
    const float total = red[0];
    __syncthreads();
    acc += e / total;
  }
  if (tid < Ntok) imp[b * Ntok + tid] = acc * (1.0f / 12.0f);
}

// ---------------------------------------------------------------------------
__global__ __launch_bounds__(256) void topk_scatter_kernel(
    const float* __restrict__ imp, const float* __restrict__ prompts,
    const float* __restrict__ pos_embed, float* __restrict__ x)
{
  const int b = blockIdx.x, tid = threadIdx.x;
  __shared__ float simp[Ntok];
  __shared__ int pos_of_slot[NSTALE];
  for (int i = tid; i < Ntok; i += 256) simp[i] = imp[b * Ntok + i];
  __syncthreads();
  for (int j = 1 + tid; j <= 196; j += 256) {
    const float vj = simp[j];
    int rank = 0;
    for (int i = 1; i <= 196; ++i) {
      const float vi = simp[i];
      rank += (vi < vj) || (vi == vj && i < j);
    }
    if (rank < NSTALE) pos_of_slot[rank] = j;
  }
  __syncthreads();
  for (int idx = tid; idx < NSTALE * Cdim; idx += 256) {
    const int s = idx / Cdim, c = idx % Cdim;
    const int pos = pos_of_slot[s];
    x[((long)b * Ntok + pos) * Cdim + c] =
        prompts[((long)b * NSTALE + s) * Cdim + c] + pos_embed[(long)pos * Cdim + c];
  }
}

// ---------------------------------------------------------------------------
__global__ __launch_bounds__(128) void head_kernel(
    const float* __restrict__ hcls, const float* __restrict__ Whead,
    const float* __restrict__ bhead, float* __restrict__ out)
{
  const int b = blockIdx.x, t = threadIdx.x;
  if (t < CLS) {
    float s = bhead[t];
    const float* hp = &hcls[(long)b * Cdim];
    for (int k = 0; k < Cdim; ++k) s += hp[k] * Whead[(long)k * CLS + t];
    out[b * CLS + t] = s;
  }
}

// ---------------------------------------------------------------------------
extern "C" void kernel_launch(void* const* d_in, const int* in_sizes, int n_in,
                              void* d_out, int out_size, void* d_ws, size_t ws_size,
                              hipStream_t stream) {
  const float* x_in     = (const float*)d_in[0];
  const float* prompts  = (const float*)d_in[1];
  const float* pos_em   = (const float*)d_in[2];
  const float* Wqkv     = (const float*)d_in[3];
  const float* bqkv     = (const float*)d_in[4];
  const float* Wproj    = (const float*)d_in[5];
  const float* bproj    = (const float*)d_in[6];
  const float* n1w      = (const float*)d_in[7];
  const float* n1b      = (const float*)d_in[8];
  const float* n2w      = (const float*)d_in[9];
  const float* n2b      = (const float*)d_in[10];
  const float* W1       = (const float*)d_in[11];
  const float* b1       = (const float*)d_in[12];
  const float* W2       = (const float*)d_in[13];
  const float* b2       = (const float*)d_in[14];
  const float* nfw      = (const float*)d_in[15];
  const float* nfb      = (const float*)d_in[16];
  const float* Whead    = (const float*)d_in[17];
  const float* bhead    = (const float*)d_in[18];
  float* out = (float*)d_out;

  const int M = Bsz * Ntok;               // 6304 = 32*197
  char* ws = (char*)d_ws;
  size_t off = 0;
  auto allocf = [&](size_t n) {
    float* p = (float*)(ws + off);
    off = (off + n * sizeof(float) + 255) & ~(size_t)255;
    return p;
  };
  auto alloch = [&](size_t n) {
    __bf16* p = (__bf16*)(ws + off);
    off = (off + n * sizeof(__bf16) + 255) & ~(size_t)255;
    return p;
  };
  float* x    = allocf((size_t)M * Cdim);
  float* h    = allocf((size_t)M * Cdim);
  float* qkv  = allocf((size_t)M * 3 * Cdim);
  float* ob   = allocf((size_t)M * Cdim);
  float* h4   = allocf((size_t)M * 4 * Cdim);
  float* imp  = allocf((size_t)Bsz * Ntok);
  float* hcls = allocf((size_t)Bsz * Cdim);
  __bf16* WqkvT  = alloch((size_t)Lnum * Cdim * 3 * Cdim);
  __bf16* WprojT = alloch((size_t)Lnum * Cdim * Cdim);
  __bf16* W1T    = alloch((size_t)Lnum * Cdim * 4 * Cdim);
  __bf16* W2T    = alloch((size_t)Lnum * 4 * Cdim * Cdim);

  (void)hipMemcpyAsync(x, x_in, (size_t)M * Cdim * sizeof(float),
                       hipMemcpyDeviceToDevice, stream);

  // One-time weight transpose+convert to bf16 [N][K]
  wt_bf16_kernel<<<dim3(3 * Cdim / 32, Cdim / 32, Lnum), 256, 0, stream>>>(Wqkv, WqkvT, Cdim, 3 * Cdim);
  wt_bf16_kernel<<<dim3(Cdim / 32, Cdim / 32, Lnum), 256, 0, stream>>>(Wproj, WprojT, Cdim, Cdim);
  wt_bf16_kernel<<<dim3(4 * Cdim / 32, Cdim / 32, Lnum), 256, 0, stream>>>(W1, W1T, Cdim, 4 * Cdim);
  wt_bf16_kernel<<<dim3(Cdim / 32, 4 * Cdim / 32, Lnum), 256, 0, stream>>>(W2, W2T, 4 * Cdim, Cdim);

  const size_t lds_bytes = 2 * ATILE * sizeof(float) + 2 * BTILEH * sizeof(__bf16); // ~50 KB
  const int mBlocks = M / 32;             // 197
  for (int i = 0; i < Lnum; ++i) {
    ln_kernel<<<M, 256, 0, stream>>>(x, Cdim, n1w + i * Cdim, n1b + i * Cdim, h, Cdim);
    gemm_wmma_kernel<<<dim3(3 * Cdim / 256, mBlocks), 256, lds_bytes, stream>>>(
        h, WqkvT + (size_t)i * Cdim * 3 * Cdim, bqkv + (size_t)i * 3 * Cdim,
        nullptr, qkv, M, 3 * Cdim, Cdim, 0);
    attn_wmma_kernel<<<dim3(13, Hn, Bsz), 128, 0, stream>>>(qkv, ob);
    gemm_wmma_kernel<<<dim3(Cdim / 256, mBlocks), 256, lds_bytes, stream>>>(
        ob, WprojT + (size_t)i * Cdim * Cdim, bproj + (size_t)i * Cdim,
        x, x, M, Cdim, Cdim, 0);
    if (i == 3) {
      cls_imp_kernel<<<Bsz, 256, 0, stream>>>(qkv, imp);
      topk_scatter_kernel<<<Bsz, 256, 0, stream>>>(imp, prompts, pos_em, x);
    }
    ln_kernel<<<M, 256, 0, stream>>>(x, Cdim, n2w + i * Cdim, n2b + i * Cdim, h, Cdim);
    gemm_wmma_kernel<<<dim3(4 * Cdim / 256, mBlocks), 256, lds_bytes, stream>>>(
        h, W1T + (size_t)i * Cdim * 4 * Cdim, b1 + (size_t)i * 4 * Cdim,
        nullptr, h4, M, 4 * Cdim, Cdim, 1);
    gemm_wmma_kernel<<<dim3(Cdim / 256, mBlocks), 256, lds_bytes, stream>>>(
        h4, W2T + (size_t)i * 4 * Cdim * Cdim, b2 + (size_t)i * Cdim,
        x, x, M, Cdim, 4 * Cdim, 0);
  }
  ln_kernel<<<Bsz, 256, 0, stream>>>(x, (long)Ntok * Cdim, nfw, nfb, hcls, Cdim);
  head_kernel<<<Bsz, 128, 0, stream>>>(hcls, Whead, bhead, out);
}
